// Cluster_7017976562037
// MI455X (gfx1250) — matrix operand order repaired and checked by reading, not compiled, not run on hardware
//
#include <hip/hip_runtime.h>
#include <hip/hip_bf16.h>

typedef __attribute__((ext_vector_type(16))) __bf16       v16bf;
typedef __attribute__((ext_vector_type(8)))  float        v8f;
typedef __attribute__((ext_vector_type(4)))  float        f4;
typedef __attribute__((ext_vector_type(4)))  unsigned int u4;

#define BSZ   32
#define SEQ   2048
#define DIM   2048
#define KC    256
#define NROWS (BSZ * SEQ)   // 65536 rows of x
#define KCH   (DIM / 32)    // 64 K-chunks of 32
#define NT    (KC / 16)     // 16 N-tiles of 16 centroids
#define CHUNK_U4 (NT * 32 * 2)          // 1024 uint4 = 16 KB per K-chunk
#define CHUNK_BYTES (CHUNK_U4 * 16)     // 16384

// ---------------------------------------------------------------------------
// CDNA5 async global -> LDS copy (ASYNCcnt-tracked, no VGPR data movement).
// lds  = wave-relative LDS byte offset (low 32 bits of the generic pointer)
// gaddr= 64-bit global byte address
// ---------------------------------------------------------------------------
__device__ __forceinline__ void async_ld_b128(unsigned lds, unsigned long long gaddr) {
    asm volatile("global_load_async_to_lds_b128 %0, %1, off"
                 :: "v"(lds), "v"(gaddr) : "memory");
}
__device__ __forceinline__ void wait_async_le4() {
    asm volatile("s_wait_asynccnt 0x4" ::: "memory");
}
__device__ __forceinline__ void wait_async_0() {
    asm volatile("s_wait_asynccnt 0x0" ::: "memory");
}

// ---------------------------------------------------------------------------
// Kernel 1: pack centers (f32, row-major [K, D]) into the bf16 B-fragment
// layout used here for v_wmma_f32_16x16x32_bf16 (contiguous K order, matching A):
//   lane % 16  -> N (column / centroid within tile)
//   lanes 0-15 -> K 0..15 of the 32-chunk, lanes 16-31 -> K 16..31
//   element e of v16bf (VGPR v = e/2, half h = e%2) -> K = kbase + e
// Stored so the GEMM kernel reads 32 contiguous bytes per lane per tile.
// ---------------------------------------------------------------------------
__global__ void pack_centers(const float* __restrict__ centers,
                             __bf16* __restrict__ pk) {
    const int chunk = blockIdx.x;        // 0..KCH-1
    const int tile  = blockIdx.y;        // 0..NT-1
    const int lane  = threadIdx.x;       // 0..31
    const int n     = tile * 16 + (lane & 15);
    const int kb    = chunk * 32 + ((lane < 16) ? 0 : 16);
    const float* src = centers + (size_t)n * DIM + kb;
    __bf16* dst = pk + (((size_t)(chunk * NT + tile) * 32) + lane) * 16;
#pragma unroll
    for (int e = 0; e < 16; ++e) dst[e] = (__bf16)src[e];
}

// ---------------------------------------------------------------------------
// Kernel 2: ||c||^2 per centroid
// ---------------------------------------------------------------------------
__global__ void center_norms(const float* __restrict__ centers,
                             float* __restrict__ cSq) {
    __shared__ float red[128];
    const int n = blockIdx.x;
    const float* c = centers + (size_t)n * DIM;
    float s = 0.f;
    for (int d = threadIdx.x; d < DIM; d += 128) { float v = c[d]; s += v * v; }
    red[threadIdx.x] = s;
    __syncthreads();
    for (int off = 64; off > 0; off >>= 1) {
        if (threadIdx.x < off) red[threadIdx.x] += red[threadIdx.x + off];
        __syncthreads();
    }
    if (threadIdx.x == 0) cSq[n] = red[0];
}

// ---------------------------------------------------------------------------
// Kernel 3: nearest-centroid labels via WMMA.
// Block = 256 threads (8 waves), 128 rows per block, 16 rows per wave.
// label = argmin_n ( cSq[n] - 2 * (x . c_n) )   (x_sq drops out of argmin)
// B fragments double-buffered in LDS via CDNA5 async global->LDS loads.
// ---------------------------------------------------------------------------
__global__ __launch_bounds__(256) void labels_kernel(
        const float*  __restrict__ x,
        const __bf16* __restrict__ pk,
        const float*  __restrict__ cSq,
        int*          __restrict__ labels) {
    __shared__ u4 sB[2][CHUNK_U4];                 // 2 x 16 KB double buffer

    const int tid     = threadIdx.x;
    const int lane    = tid & 31;
    const int wave    = tid >> 5;
    const int half    = lane >> 4;                 // 0: lanes 0-15, 1: lanes 16-31
    const int lcol    = lane & 15;
    const int rowBase = blockIdx.x * 128 + wave * 16;

    // Per-lane column norms: lane handles columns n = t*16 + lcol
    float cs[NT];
#pragma unroll
    for (int t = 0; t < NT; ++t) cs[t] = cSq[t * 16 + lcol];

    v8f acc[NT] = {};                              // 16 tiles of 16x16 f32 accum

    // A-fragment addressing (contiguous K order, matching packed B; 16-bit A 16x32):
    //   lane row M = lane%16; kbase = (lane<16 ? 0 : 16)
    //   elements e=0..15 -> K = kbase + e
    const int kb = half ? 16 : 0;
    const float* xr = x + (size_t)(rowBase + lcol) * DIM + kb;

    // LDS byte offsets of the two staging buffers for this thread's 16B slot.
    // Flat LDS addressing keeps the wave-relative offset in addr[31:0].
    const unsigned lds0 = (unsigned)(unsigned long long)(const void*)&sB[0][0]
                        + (unsigned)tid * 16u;
    const unsigned lds1 = lds0 + (unsigned)CHUNK_BYTES;
    const unsigned long long gpk = (unsigned long long)pk + (unsigned long long)tid * 16u;

    // Prefetch chunk 0 into buffer 0 (4 async b128 per thread = 16 KB/block)
    {
        const unsigned long long g = gpk;
        async_ld_b128(lds0,          g);
        async_ld_b128(lds0 + 4096u,  g + 4096ull);
        async_ld_b128(lds0 + 8192u,  g + 8192ull);
        async_ld_b128(lds0 + 12288u, g + 12288ull);
    }

    for (int c = 0; c < KCH; ++c) {
        if (c + 1 < KCH) {
            // Issue next chunk into the other buffer, then wait only for the
            // 4 older (in-order) async loads of the current chunk.
            const unsigned nl = (c & 1) ? lds0 : lds1;
            const unsigned long long g = gpk + (unsigned long long)(c + 1) * CHUNK_BYTES;
            async_ld_b128(nl,          g);
            async_ld_b128(nl + 4096u,  g + 4096ull);
            async_ld_b128(nl + 8192u,  g + 8192ull);
            async_ld_b128(nl + 12288u, g + 12288ull);
            wait_async_le4();
        } else {
            wait_async_0();
        }
        __syncthreads();                           // chunk c visible to all waves

        // Build A fragment: 16 f32 -> bf16 (streaming read-once: NT loads)
        const float* xp = xr + c * 32;
        f4 p0 = __builtin_nontemporal_load((const f4*)(xp));
        f4 p1 = __builtin_nontemporal_load((const f4*)(xp + 4));
        f4 p2 = __builtin_nontemporal_load((const f4*)(xp + 8));
        f4 p3 = __builtin_nontemporal_load((const f4*)(xp + 12));
        v16bf a;
        a[0]  = (__bf16)p0.x; a[1]  = (__bf16)p0.y; a[2]  = (__bf16)p0.z; a[3]  = (__bf16)p0.w;
        a[4]  = (__bf16)p1.x; a[5]  = (__bf16)p1.y; a[6]  = (__bf16)p1.z; a[7]  = (__bf16)p1.w;
        a[8]  = (__bf16)p2.x; a[9]  = (__bf16)p2.y; a[10] = (__bf16)p2.z; a[11] = (__bf16)p2.w;
        a[12] = (__bf16)p3.x; a[13] = (__bf16)p3.y; a[14] = (__bf16)p3.z; a[15] = (__bf16)p3.w;

        const u4* sBc = sB[c & 1];
#pragma unroll
        for (int t = 0; t < NT; ++t) {
            union { u4 u[2]; v16bf v; } fb;
            fb.u[0] = sBc[(t * 32 + lane) * 2 + 0];
            fb.u[1] = sBc[(t * 32 + lane) * 2 + 1];
            acc[t] = __builtin_amdgcn_wmma_f32_16x16x32_bf16(
                /*neg_a=*/false, a, /*neg_b=*/false, fb.v,
                /*c_mod=*/(short)0, acc[t], /*reuse_a=*/false, /*reuse_b=*/false);
        }
        __syncthreads();  // all waves done reading before this buffer is re-filled
    }

    // Argmin. C/D layout: VGPR r holds row M = r + 8*half; column N = t*16 + lcol.
    float bestv[8];
    int   bestn[8];
#pragma unroll
    for (int r = 0; r < 8; ++r) { bestv[r] = 3.4e38f; bestn[r] = 0; }
#pragma unroll
    for (int t = 0; t < NT; ++t) {
        const int n = t * 16 + lcol;
#pragma unroll
        for (int r = 0; r < 8; ++r) {
            const float v = cs[t] - 2.0f * acc[t][r];
            if (v < bestv[r]) { bestv[r] = v; bestn[r] = n; }
        }
    }
    // Reduce across the 16 lanes of this half (xor 1,2,4,8 stays within the half)
#pragma unroll
    for (int r = 0; r < 8; ++r) {
#pragma unroll
        for (int s = 1; s < 16; s <<= 1) {
            const float ov = __shfl_xor(bestv[r], s, 32);
            const int   on = __shfl_xor(bestn[r], s, 32);
            if (ov < bestv[r] || (ov == bestv[r] && on < bestn[r])) {
                bestv[r] = ov; bestn[r] = on;
            }
        }
    }
    if (lcol == 0) {
#pragma unroll
        for (int r = 0; r < 8; ++r)
            labels[rowBase + half * 8 + r] = bestn[r];
    }
}

// ---------------------------------------------------------------------------
// Kernel 4: state_weight[b][i][j] = (label[b][i] == label[b][j]) ? 1 : 0
// 16-row x 2048-col tiles; labels staged in LDS; nontemporal f32x4 stores
// (537 MB stream >> 192 MB L2 -> bypass).
// ---------------------------------------------------------------------------
__global__ __launch_bounds__(256) void mask_kernel(
        const int* __restrict__ labels, float* __restrict__ out) {
    const int b   = blockIdx.x >> 7;       // 32 batches
    const int it  = blockIdx.x & 127;      // 128 row-tiles of 16
    const int tid = threadIdx.x;
    __shared__ int lj[SEQ];
    __shared__ int li[16];
    const int* lb = labels + b * SEQ;
#pragma unroll
    for (int m = 0; m < 8; ++m) lj[tid + m * 256] = lb[tid + m * 256];
    if (tid < 16) li[tid] = lb[it * 16 + tid];
    __syncthreads();

    const int c0 = tid * 8;                // 8 consecutive columns per thread
    int jl[8];
#pragma unroll
    for (int q = 0; q < 8; ++q) jl[q] = lj[c0 + q];

#pragma unroll
    for (int r = 0; r < 16; ++r) {
        const int lv = li[r];
        f4 m0, m1;
        m0.x = (jl[0] == lv) ? 1.f : 0.f; m0.y = (jl[1] == lv) ? 1.f : 0.f;
        m0.z = (jl[2] == lv) ? 1.f : 0.f; m0.w = (jl[3] == lv) ? 1.f : 0.f;
        m1.x = (jl[4] == lv) ? 1.f : 0.f; m1.y = (jl[5] == lv) ? 1.f : 0.f;
        m1.z = (jl[6] == lv) ? 1.f : 0.f; m1.w = (jl[7] == lv) ? 1.f : 0.f;
        float* row = out + ((size_t)(b * SEQ + it * 16 + r)) * SEQ + c0;
        __builtin_nontemporal_store(m0, (f4*)row);
        __builtin_nontemporal_store(m1, (f4*)(row + 4));
    }
}

// ---------------------------------------------------------------------------
// Kernel 5: labels as float into output tail
// ---------------------------------------------------------------------------
__global__ void labels_f32(const int* __restrict__ labels, float* __restrict__ out) {
    const int i = blockIdx.x * 256 + threadIdx.x;
    out[i] = (float)labels[i];
}

// ---------------------------------------------------------------------------
extern "C" void kernel_launch(void* const* d_in, const int* in_sizes, int n_in,
                              void* d_out, int out_size, void* d_ws, size_t ws_size,
                              hipStream_t stream) {
    const float* x       = (const float*)d_in[0];   // [32, 2048, 2048] f32
    const float* centers = (const float*)d_in[1];   // [256, 2048] f32
    float* out = (float*)d_out;                     // [mask 32*2048*2048][labels 65536]

    char* ws = (char*)d_ws;
    int*    labels = (int*)ws;                      // 256 KB
    float*  cSq    = (float*)(ws + (256 << 10));    // 1 KB
    __bf16* pk     = (__bf16*)(ws + (512 << 10));   // 1 MB packed bf16 B fragments

    pack_centers<<<dim3(KCH, NT), 32, 0, stream>>>(centers, pk);
    center_norms<<<KC, 128, 0, stream>>>(centers, cSq);
    labels_kernel<<<NROWS / 128, 256, 0, stream>>>(x, pk, cSq, labels);
    mask_kernel<<<BSZ * (SEQ / 16), 256, 0, stream>>>(labels, out);
    labels_f32<<<NROWS / 256, 256, 0, stream>>>(labels, out + (size_t)BSZ * SEQ * SEQ);
}
